// GAT_38946763440878
// MI455X (gfx1250) — compile-verified, matrix operand order checked by reading, not run on hardware
//
#include <hip/hip_runtime.h>
#include <hip/hip_bf16.h>
#include <stdint.h>

#define D_FEAT 768
#define NEG_SLOPE 0.2f

typedef __attribute__((ext_vector_type(16))) __bf16 v16bf;
typedef __attribute__((ext_vector_type(8)))  __bf16 v8bf;
typedef __attribute__((ext_vector_type(8)))  float  v8f;

// ---------- helpers ----------
__device__ __forceinline__ unsigned short f2bf(float x) {
    unsigned int u = __float_as_uint(x);
    unsigned int r = u + 0x7FFFu + ((u >> 16) & 1u);   // round-to-nearest-even
    return (unsigned short)(r >> 16);
}

__device__ __forceinline__ float atomicMaxFloat(float* addr, float value) {
    if (value >= 0.0f)
        return __int_as_float(atomicMax((int*)addr, __float_as_int(value)));
    else
        return __uint_as_float(atomicMin((unsigned int*)addr, __float_as_uint(value)));
}

// 16-byte global -> LDS copy; async on gfx1250 (ASYNCcnt), sync fallback otherwise.
// VGLOBAL async encoding: VDST = LDS byte offset VGPR, VADDR = 64-bit global addr, GV mode.
__device__ __forceinline__ void copy16_g2l(const unsigned short* g, unsigned short* l) {
#if defined(__gfx1250__)
    unsigned int lds_off = (unsigned int)(unsigned long long)(const void*)l; // low 32b = LDS offset
    asm volatile("global_load_async_to_lds_b128 %0, %1, off"
                 :
                 : "v"(lds_off), "v"((unsigned long long)(const void*)g)
                 : "memory");
#else
    *(int4*)l = *(const int4*)g;
#endif
}

__device__ __forceinline__ void wait_async_all() {
#if defined(__gfx1250__)
 #if __has_builtin(__builtin_amdgcn_s_wait_asynccnt)
    __builtin_amdgcn_s_wait_asynccnt(0);
 #else
    asm volatile("s_wait_asynccnt 0" ::: "memory");
 #endif
#endif
}

__device__ __forceinline__ v16bf load_frag(const unsigned short* p) {
    // two 16B loads: K-block [0..7] and K-block [16..23] for this lane-half
    v8bf lo = *(const v8bf*)(p);
    v8bf hi = *(const v8bf*)(p + 16);
    return __builtin_shufflevector(lo, hi, 0,1,2,3,4,5,6,7,8,9,10,11,12,13,14,15);
}

// ---------- conversion kernels ----------
__global__ void k_conv_w_transpose(const float* __restrict__ W, unsigned short* __restrict__ Wt, int D) {
    int idx = blockIdx.x * blockDim.x + threadIdx.x;
    if (idx >= D * D) return;
    int k = idx / D, n = idx % D;               // coalesced read of W[k][n]
    Wt[(size_t)n * D + k] = f2bf(W[idx]);       // Wt[n][k] = W[k][n]
}

__global__ void k_conv_x(const float* __restrict__ X, unsigned short* __restrict__ Xb,
                         int N, int Mpad, int D) {
    int idx = blockIdx.x * blockDim.x + threadIdx.x;
    if (idx >= Mpad * D) return;
    int r = idx / D;
    Xb[idx] = (r < N) ? f2bf(X[idx]) : (unsigned short)0;
}

__global__ void k_finalize_relu_bf16(const float* __restrict__ acc, const float* __restrict__ b,
                                     unsigned short* __restrict__ Xb, int N, int Mpad, int D) {
    int idx = blockIdx.x * blockDim.x + threadIdx.x;
    if (idx >= Mpad * D) return;
    int r = idx / D, c = idx % D;
    float v = 0.0f;
    if (r < N) {
        v = acc[(size_t)r * D + c] + b[c];
        v = v > 0.0f ? v : 0.0f;
    }
    Xb[idx] = f2bf(v);
}

__global__ void k_fill(float* __restrict__ p, float v, long long n) {
    long long i = (long long)blockIdx.x * blockDim.x + threadIdx.x;
    if (i < n) p[i] = v;
}

__global__ void k_add_bias(float* __restrict__ out, const float* __restrict__ b, int N, int D) {
    int idx = blockIdx.x * blockDim.x + threadIdx.x;
    if (idx >= N * D) return;
    out[idx] += b[idx % D];
}

// ---------- WMMA GEMM with async-LDS double buffering ----------
// C[M x Dout] = A(bf16, M x K) * Wt(bf16, Dout x K)^T
// Block: 256 thr = 8 waves (4M x 2N). Wave tile 32x64 (8 WMMA accums).
// Block tile 128x128, K-step 32. LDS: double-buffered A/B slabs, rows padded to 80B.
#define ROWP 40   // padded row stride in ushorts (80B, 16B aligned)

__device__ __forceinline__ void copy_slab(const unsigned short* __restrict__ G, int K,
                                          int r0, int k0, unsigned short* dst, int t) {
    // 128 rows x 32 ushorts; 256 threads x 2 segments of 16B
    int row = t >> 2;            // 0..63
    int seg = t & 3;             // 4 x 16B per row
    const unsigned short* g0 = G + (size_t)(r0 + row) * K + k0 + seg * 8;
    copy16_g2l(g0,                  dst + row * ROWP + seg * 8);
    copy16_g2l(g0 + (size_t)64 * K, dst + (row + 64) * ROWP + seg * 8);
}

__global__ __launch_bounds__(256) void k_gemm_bf16(const unsigned short* __restrict__ A,
                                                   const unsigned short* __restrict__ B,
                                                   float* __restrict__ C,
                                                   int K, int Dout) {
    __shared__ unsigned short sA[2][128 * ROWP];
    __shared__ unsigned short sB[2][128 * ROWP];

    const int t     = threadIdx.x;
    const int lane  = t & 31;
    const int wave  = t >> 5;
    const int waveM = wave & 3;       // 4 M-waves * 32 rows
    const int waveN = wave >> 2;      // 2 N-waves * 64 cols
    const int half  = lane >> 4;      // K sub-block selector (ISA A/B frag layout)
    const int row16 = lane & 15;

    const int m0 = blockIdx.x * 128;
    const int n0 = blockIdx.y * 128;
    const int KT = K >> 5;

    // prologue: fill buffer 0
    copy_slab(A, K, m0, 0, sA[0], t);
    copy_slab(B, K, n0, 0, sB[0], t);
    wait_async_all();
    __syncthreads();

    v8f acc[2][4];
#pragma unroll
    for (int i = 0; i < 2; ++i)
#pragma unroll
        for (int j = 0; j < 4; ++j) acc[i][j] = (v8f){};

    for (int kt = 0; kt < KT; ++kt) {
        const int cur = kt & 1;
        const bool more = (kt + 1 < KT);
        if (more) {   // overlap next slab fill with compute
            copy_slab(A, K, m0, (kt + 1) << 5, sA[cur ^ 1], t);
            copy_slab(B, K, n0, (kt + 1) << 5, sB[cur ^ 1], t);
        }

        const unsigned short* aBase = sA[cur] + (waveM * 32 + row16) * ROWP + half * 8;
        v16bf a0 = load_frag(aBase);
        v16bf a1 = load_frag(aBase + 16 * ROWP);
        const unsigned short* bBase = sB[cur] + (waveN * 64 + row16) * ROWP + half * 8;
#pragma unroll
        for (int j = 0; j < 4; ++j) {
            v16bf b = load_frag(bBase + j * 16 * ROWP);
            acc[0][j] = __builtin_amdgcn_wmma_f32_16x16x32_bf16(false, a0, false, b,
                                                                (short)0, acc[0][j], false, false);
            acc[1][j] = __builtin_amdgcn_wmma_f32_16x16x32_bf16(false, a1, false, b,
                                                                (short)0, acc[1][j], false, false);
        }

        if (more) {
            wait_async_all();     // own async ops for next slab done
            __syncthreads();      // everyone finished reading cur, next slab visible
        }
    }

    // C/D layout: VGPR r -> M = r + 8*half ; lane[3:0] -> N
#pragma unroll
    for (int i = 0; i < 2; ++i)
#pragma unroll
        for (int j = 0; j < 4; ++j) {
            float* cp = C + (size_t)(m0 + waveM * 32 + i * 16 + half * 8) * Dout
                          + n0 + waveN * 64 + j * 16 + row16;
#pragma unroll
            for (int r = 0; r < 8; ++r)
                cp[(size_t)r * Dout] = acc[i][j][r];
        }
}

// ---------- attention: per-node scores ----------
__global__ void k_node_scores(const float* __restrict__ H, const float* __restrict__ a_src,
                              const float* __restrict__ a_dst, float* __restrict__ es,
                              float* __restrict__ ed, int D) {
    __shared__ float s1[256], s2[256];
    const int n = blockIdx.x;
    const int t = threadIdx.x;
    const float* h = H + (size_t)n * D;
    float p1 = 0.0f, p2 = 0.0f;
    for (int c = t; c < D; c += 256) {
        float v = h[c];
        p1 += v * a_src[c];
        p2 += v * a_dst[c];
    }
    s1[t] = p1; s2[t] = p2;
    __syncthreads();
    for (int st = 128; st > 0; st >>= 1) {
        if (t < st) { s1[t] += s1[t + st]; s2[t] += s2[t + st]; }
        __syncthreads();
    }
    if (t == 0) { es[n] = s1[0]; ed[n] = s2[0]; }
}

// ---------- attention: edge logits + segment max ----------
__global__ void k_edge_logits(const int* __restrict__ ei, const float* __restrict__ es,
                              const float* __restrict__ ed, float* __restrict__ e_edge,
                              float* __restrict__ m, int E, int E2) {
    int e = blockIdx.x * blockDim.x + threadIdx.x;
    if (e >= E2) return;
    int s = (e < E) ? ei[e]     : (e - E);   // self-loop tail
    int d = (e < E) ? ei[E + e] : (e - E);
    float v = es[s] + ed[d];
    v = v > 0.0f ? v : NEG_SLOPE * v;        // leaky relu
    e_edge[e] = v;
    atomicMaxFloat(&m[d], v);
}

// ---------- attention: exp + segment sum ----------
__global__ void k_edge_exp(const int* __restrict__ ei, const float* __restrict__ e_edge,
                           const float* __restrict__ m, float* __restrict__ w_edge,
                           float* __restrict__ z, int E, int E2) {
    int e = blockIdx.x * blockDim.x + threadIdx.x;
    if (e >= E2) return;
    int d = (e < E) ? ei[E + e] : (e - E);
    float w = expf(e_edge[e] - m[d]);
    w_edge[e] = w;
    atomicAdd(&z[d], w);
}

// ---------- attention: alpha-weighted scatter ----------
__global__ void k_scatter(const float* __restrict__ H, const float* __restrict__ w_edge,
                          const float* __restrict__ z, const int* __restrict__ ei,
                          float* __restrict__ out, int E, int D) {
    int e = blockIdx.x;
    int s = (e < E) ? ei[e]     : (e - E);
    int d = (e < E) ? ei[E + e] : (e - E);
    float alpha = w_edge[e] / z[d];
    const float* hs = H + (size_t)s * D;
    float* od = out + (size_t)d * D;
    for (int c = threadIdx.x; c < D; c += blockDim.x)
        atomicAdd(&od[c], hs[c] * alpha);
}

// ---------- host launcher ----------
static inline size_t align256(size_t x) { return (x + 255) & ~(size_t)255; }

extern "C" void kernel_launch(void* const* d_in, const int* in_sizes, int n_in,
                              void* d_out, int out_size, void* d_ws, size_t ws_size,
                              hipStream_t stream) {
    const int D = D_FEAT;
    const int N = in_sizes[0] / D;       // 10000
    const int E = in_sizes[1] / 2;       // 100000
    const int E2 = E + N;                // + self loops
    const int Mpad = ((N + 127) / 128) * 128;

    const float* x   = (const float*)d_in[0];
    const int*   ei  = (const int*)d_in[1];
    const float* W1  = (const float*)d_in[2];
    const float* as1 = (const float*)d_in[3];
    const float* ad1 = (const float*)d_in[4];
    const float* b1  = (const float*)d_in[5];
    const float* W2  = (const float*)d_in[6];
    const float* as2 = (const float*)d_in[7];
    const float* ad2 = (const float*)d_in[8];
    const float* b2  = (const float*)d_in[9];
    float* out = (float*)d_out;

    // workspace carve-up
    char* ws = (char*)d_ws;
    size_t off = 0;
    unsigned short* Wt1 = (unsigned short*)(ws + off); off += align256((size_t)D * D * 2);
    unsigned short* Wt2 = (unsigned short*)(ws + off); off += align256((size_t)D * D * 2);
    unsigned short* Xb  = (unsigned short*)(ws + off); off += align256((size_t)Mpad * D * 2);
    float* H    = (float*)(ws + off); off += align256((size_t)Mpad * D * 4);
    float* acc1 = (float*)(ws + off); off += align256((size_t)N * D * 4);
    float* es   = (float*)(ws + off); off += align256((size_t)N * 4);
    float* ed   = (float*)(ws + off); off += align256((size_t)N * 4);
    float* mseg = (float*)(ws + off); off += align256((size_t)N * 4);
    float* zseg = (float*)(ws + off); off += align256((size_t)N * 4);
    float* e_edge = (float*)(ws + off); off += align256((size_t)E2 * 4);
    float* w_edge = (float*)(ws + off); off += align256((size_t)E2 * 4);
    if (off > ws_size) return;

    const int TPB = 256;
    const int gDD  = (D * D + TPB - 1) / TPB;
    const int gPad = (Mpad * D + TPB - 1) / TPB;
    const int gND  = (N * D + TPB - 1) / TPB;
    const int gE2  = (E2 + TPB - 1) / TPB;
    const int gN   = (N + TPB - 1) / TPB;
    dim3 gGemm(Mpad / 128, D / 128);

    // weights -> bf16 transposed (once)
    k_conv_w_transpose<<<gDD, TPB, 0, stream>>>(W1, Wt1, D);
    k_conv_w_transpose<<<gDD, TPB, 0, stream>>>(W2, Wt2, D);

    // ===== layer 1 =====
    k_conv_x<<<gPad, TPB, 0, stream>>>(x, Xb, N, Mpad, D);
    k_gemm_bf16<<<gGemm, TPB, 0, stream>>>(Xb, Wt1, H, D, D);
    k_node_scores<<<N, TPB, 0, stream>>>(H, as1, ad1, es, ed, D);
    k_fill<<<gN, TPB, 0, stream>>>(mseg, -INFINITY, N);
    k_fill<<<gN, TPB, 0, stream>>>(zseg, 0.0f, N);
    k_fill<<<gND, TPB, 0, stream>>>(acc1, 0.0f, (long long)N * D);
    k_edge_logits<<<gE2, TPB, 0, stream>>>(ei, es, ed, e_edge, mseg, E, E2);
    k_edge_exp<<<gE2, TPB, 0, stream>>>(ei, e_edge, mseg, w_edge, zseg, E, E2);
    k_scatter<<<E2, TPB, 0, stream>>>(H, w_edge, zseg, ei, acc1, E, D);

    // ===== layer 2 =====
    k_finalize_relu_bf16<<<gPad, TPB, 0, stream>>>(acc1, b1, Xb, N, Mpad, D);
    k_gemm_bf16<<<gGemm, TPB, 0, stream>>>(Xb, Wt2, H, D, D);
    k_node_scores<<<N, TPB, 0, stream>>>(H, as2, ad2, es, ed, D);
    k_fill<<<gN, TPB, 0, stream>>>(mseg, -INFINITY, N);
    k_fill<<<gN, TPB, 0, stream>>>(zseg, 0.0f, N);
    k_fill<<<gND, TPB, 0, stream>>>(out, 0.0f, (long long)N * D);
    k_edge_logits<<<gE2, TPB, 0, stream>>>(ei, es, ed, e_edge, mseg, E, E2);
    k_edge_exp<<<gE2, TPB, 0, stream>>>(ei, e_edge, mseg, w_edge, zseg, E, E2);
    k_scatter<<<E2, TPB, 0, stream>>>(H, w_edge, zseg, ei, out, E, D);
    k_add_bias<<<gND, TPB, 0, stream>>>(out, b2, N, D);
}
